// Cross_Attention_55671366091237
// MI455X (gfx1250) — matrix operand order repaired and checked by reading, not compiled, not run on hardware
//
#include <hip/hip_runtime.h>

// MI455X / gfx1250, wave32, WMMA f32_16x16x32_f16 everywhere.
// Pipeline: fc(x), fc(y) -> qkv+modulate -> flash attention -> out GEMM.
// Workspace use: oq,okv f32 (4MB) + Q,K,V,attn f16 (32MB) = ~36MB.
// CDNA5 data movement: global_load_async_to_lds_b128 (ASYNCcnt) for the two
// pure-f16 staging sites + global_prefetch_b8 for next-tile warming.

typedef __attribute__((ext_vector_type(16))) _Float16 v16h;
typedef __attribute__((ext_vector_type(8)))  float    v8f;

#define LDA 40   // halfs; 80B row stride -> 16B-aligned rows for async b128
#define LDB 34
#define LDV 68

__device__ __forceinline__ v8f wmma_f16(v16h a, v16h b, v8f c) {
  // (neg_a, A, neg_b, B, c_mod, C, reuse_a, reuse_b)
  return __builtin_amdgcn_wmma_f32_16x16x32_f16(false, a, false, b, (short)0, c,
                                                false, false);
}

__device__ __forceinline__ float bpermf(float v, int srcLane) {
  return __builtin_bit_cast(
      float, __builtin_amdgcn_ds_bpermute(srcLane << 2,
                                          __builtin_bit_cast(int, v)));
}

// Async DMA: 16 bytes global -> LDS, GVS addressing (uniform SGPR base +
// per-lane 32-bit byte offset). Tracked by ASYNCcnt.
__device__ __forceinline__ void async_b128(unsigned lds_addr,
                                           unsigned long long sbase,
                                           unsigned voff) {
  asm volatile("global_load_async_to_lds_b128 %0, %1, %2"
               :
               : "v"(lds_addr), "v"(voff), "s"(sbase)
               : "memory");
}

__device__ __forceinline__ void wait_async() {
  asm volatile("s_wait_asynccnt 0x0" ::: "memory");
}

__device__ __forceinline__ unsigned lds_off(const void* p) {
  // flat pointers into the LDS aperture carry the LDS byte offset in [31:0]
  return (unsigned)(unsigned long long)p;
}

// A fragment (16 rows x 32 k, f16) from row-major LDS tile with leading dim ld.
// ISA layout: lane L holds row M=L%16; VGPR v pair covers k = (v<4 ? 2v : 8+2v) + 8*(L/16).
__device__ __forceinline__ v16h frag_a(const _Float16* tile, int lane, int ld) {
  const _Float16* rp = tile + (lane & 15) * ld;
  const int hf = (lane >> 4) & 1;
  v16h a;
#pragma unroll
  for (int v = 0; v < 8; ++v) {
    int k0 = ((v < 4) ? (2 * v) : (8 + 2 * v)) + 8 * hf;
    a[2 * v]     = rp[k0];
    a[2 * v + 1] = rp[k0 + 1];
  }
  return a;
}

// B fragment (32 k x 16 n) from LDS tile stored transposed: element (n,k) at n*ld+k.
// lane L holds n=L%16; VGPR v pair covers k = (v<4 ? 2v : 8+2v) + 8*(L/16) (same as A).
__device__ __forceinline__ v16h frag_b(const _Float16* tile, int lane, int ld) {
  const _Float16* rp = tile + (lane & 15) * ld;
  const int hf = (lane >> 4) & 1;
  v16h b;
#pragma unroll
  for (int v = 0; v < 8; ++v) {
    int k0 = ((v < 4) ? (2 * v) : (8 + 2 * v)) + 8 * hf;
    b[2 * v]     = rp[k0];
    b[2 * v + 1] = rp[k0 + 1];
  }
  return b;
}

// Stage a 128x32 A tile (convert f32->f16) into LDS, 256 threads.
__device__ __forceinline__ void stage_a_f32(_Float16* As, const float* A,
                                            int lda, int rowBase, int kBase,
                                            int t) {
  const int row = t >> 1;
  const int col = (t & 1) * 16;
  const float* src = A + (size_t)(rowBase + row) * lda + kBase + col;
  _Float16* dst = As + row * LDA + col;
#pragma unroll
  for (int j = 0; j < 16; ++j) dst[j] = (_Float16)src[j];
}

// Stage NTx32 B tile transposed ([n][k]) into LDS from row-major f32 W[k][n].
template <int NT>
__device__ __forceinline__ void stage_bt(_Float16* Bt, const float* B, int ldb,
                                         int nBase, int kBase, int t) {
  constexpr int PER = NT * 32 / 256;
#pragma unroll
  for (int i = 0; i < PER; ++i) {
    int j = t + i * 256;
    int n = j % NT;
    int k = j / NT;
    Bt[n * LDB + k] = (_Float16)B[(size_t)(kBase + k) * ldb + nBase + n];
  }
}

// ---------------- fc: Out[8192,64] = X[8192,512] @ Wfc[512,64] + bfc ----------
__global__ __launch_bounds__(256) void fc_kernel(const float* __restrict__ X,
                                                 const float* __restrict__ Wfc,
                                                 const float* __restrict__ bfc,
                                                 float* __restrict__ Out) {
  __shared__ _Float16 As[128 * LDA];
  __shared__ _Float16 Bt[64 * LDB];
  const int t = threadIdx.x, lane = t & 31, wave = t >> 5;
  const int wm = wave & 3, wn = wave >> 2;  // 4x2 wave grid, 32x32 per wave
  const int rowBase = blockIdx.x * 128;
  v8f acc[2][2] = {};
  for (int kt = 0; kt < 16; ++kt) {
    __syncthreads();
    stage_a_f32(As, X, 512, rowBase, kt * 32, t);
    stage_bt<64>(Bt, Wfc, 64, 0, kt * 32, t);
    __syncthreads();
    v16h a0 = frag_a(As + (wm * 32) * LDA, lane, LDA);
    v16h a1 = frag_a(As + (wm * 32 + 16) * LDA, lane, LDA);
#pragma unroll
    for (int nf = 0; nf < 2; ++nf) {
      v16h b = frag_b(Bt + (wn * 32 + nf * 16) * LDB, lane, LDB);
      acc[0][nf] = wmma_f16(a0, b, acc[0][nf]);
      acc[1][nf] = wmma_f16(a1, b, acc[1][nf]);
    }
  }
  const int hf = lane >> 4, ln = lane & 15;
#pragma unroll
  for (int mf = 0; mf < 2; ++mf)
#pragma unroll
    for (int nf = 0; nf < 2; ++nf)
#pragma unroll
      for (int i = 0; i < 8; ++i) {
        int row = rowBase + wm * 32 + mf * 16 + i + 8 * hf;
        int col = wn * 32 + nf * 16 + ln;
        Out[(size_t)row * 64 + col] = acc[mf][nf][i] + bfc[col];
      }
}

// --- qkv: x @ W_qkv, modulated by oq/okv, scattered to Q/K/V f16 [B,H,N,64] ---
__global__ __launch_bounds__(256) void qkv_kernel(
    const float* __restrict__ X, const float* __restrict__ Wqkv,
    const float* __restrict__ OQ, const float* __restrict__ OKV,
    _Float16* __restrict__ Qh, _Float16* __restrict__ Kh,
    _Float16* __restrict__ Vh) {
  __shared__ _Float16 As[128 * LDA];
  __shared__ _Float16 Bt[128 * LDB];
  const int t = threadIdx.x, lane = t & 31, wave = t >> 5;
  const int wm = wave & 3, wn = wave >> 2;  // 4x2 waves, 32x64 per wave
  const int rowBase = blockIdx.x * 128;
  const int colBase = blockIdx.y * 128;
  v8f acc[2][4] = {};
  for (int kt = 0; kt < 16; ++kt) {
    __syncthreads();
    stage_a_f32(As, X, 512, rowBase, kt * 32, t);
    stage_bt<128>(Bt, Wqkv, 1536, colBase, kt * 32, t);
    __syncthreads();
    v16h a0 = frag_a(As + (wm * 32) * LDA, lane, LDA);
    v16h a1 = frag_a(As + (wm * 32 + 16) * LDA, lane, LDA);
#pragma unroll
    for (int nf = 0; nf < 4; ++nf) {
      v16h b = frag_b(Bt + (wn * 64 + nf * 16) * LDB, lane, LDB);
      acc[0][nf] = wmma_f16(a0, b, acc[0][nf]);
      acc[1][nf] = wmma_f16(a1, b, acc[1][nf]);
    }
  }
  const int hf = lane >> 4, ln = lane & 15;
#pragma unroll
  for (int mf = 0; mf < 2; ++mf)
#pragma unroll
    for (int nf = 0; nf < 4; ++nf)
#pragma unroll
      for (int i = 0; i < 8; ++i) {
        int row = rowBase + wm * 32 + mf * 16 + i + 8 * hf;  // 0..8191
        int col = colBase + wn * 64 + nf * 16 + ln;          // 0..1535
        int s = col >> 9;  // 0=q 1=k 2=v
        int cc = col & 511;
        int h = cc >> 6, d = cc & 63;
        float mod = ((s == 0) ? OQ : OKV)[(size_t)row * 64 + d];
        _Float16* dst = (s == 0) ? Qh : ((s == 1) ? Kh : Vh);
        int b = row >> 11, n = row & 2047;
        dst[(((size_t)(b * 8 + h) * 2048 + n) << 6) + d] =
            (_Float16)(acc[mf][nf][i] * mod);
      }
}

// ---------------- flash attention per (b,h), 128 q rows per block ------------
__global__ __launch_bounds__(256) void attn_kernel(
    const _Float16* __restrict__ Qh, const _Float16* __restrict__ Kh,
    const _Float16* __restrict__ Vh, _Float16* __restrict__ Oh) {
  __shared__ __align__(16) _Float16 Ks[64 * 64];  // [key][d] == Bt for QK^T
  __shared__ _Float16 Vts[64 * LDV];              // [d][key] == Bt for PV
  __shared__ float    Ssm[8][16 * 64];            // per-wave scores
  __shared__ _Float16 Psm[8][16 * 64];            // per-wave probabilities
  const int t = threadIdx.x, lane = t & 31, wave = t >> 5;
  const int ln = lane & 15, hf = lane >> 4;
  const int h = blockIdx.y, b = blockIdx.z;
  const int bh = b * 8 + h;
  const _Float16* Qb = Qh + ((size_t)bh * 2048) * 64;
  const _Float16* Kb = Kh + ((size_t)bh * 2048) * 64;
  const _Float16* Vb = Vh + ((size_t)bh * 2048) * 64;
  const unsigned long long kbase_u = (unsigned long long)Kb;
  const int qrow0 = blockIdx.x * 128 + wave * 16;

  // Q fragments (d 0..31 and 32..63) straight from global (A layout)
  v16h qa0, qa1;
  {
    const _Float16* rp = Qb + (size_t)(qrow0 + ln) * 64;
#pragma unroll
    for (int v = 0; v < 8; ++v) {
      int k0 = ((v < 4) ? (2 * v) : (8 + 2 * v)) + 8 * hf;
      qa0[2 * v] = rp[k0];      qa0[2 * v + 1] = rp[k0 + 1];
      qa1[2 * v] = rp[32 + k0]; qa1[2 * v + 1] = rp[32 + k0 + 1];
    }
  }

  float m = -1e30f, l = 0.f;
  const float scale = 0.125f;  // 1/sqrt(64)
  v8f o[4] = {};
  float* Srow = &Ssm[wave][ln * 64 + hf * 32];
  _Float16* Prow = &Psm[wave][ln * 64 + hf * 32];

  for (int kt = 0; kt < 32; ++kt) {
    __syncthreads();
    {  // async DMA: K tile [64 keys][64 d], 32B per thread (2 x b128)
      int key = t >> 2, c0 = (t & 3) * 16;
      unsigned dst = lds_off(&Ks[key * 64 + c0]);
      unsigned voff = (unsigned)(((kt * 64 + key) * 64 + c0) * 2);
      async_b128(dst, kbase_u, voff);
      async_b128(dst + 16, kbase_u, voff + 16);
    }
#pragma unroll
    for (int i = 0; i < 16; ++i) {  // stage V transposed [d][key] (VALU path)
      int j = t + i * 256;
      int d = j & 63, key = j >> 6;
      Vts[d * LDV + key] = Vb[(size_t)(kt * 64 + key) * 64 + d];
    }
    if (kt + 1 < 32) {  // warm next K/V tiles
      __builtin_prefetch(Kb + (size_t)((kt + 1) * 64 + (t >> 2)) * 64, 0, 0);
      __builtin_prefetch(Vb + (size_t)((kt + 1) * 64 + (t >> 2)) * 64, 0, 0);
    }
    wait_async();
    __syncthreads();

    // S(16x64) = Q(16x64) . K^T : reduce over d (two 32-wide WMMA steps)
    v8f s[4];
#pragma unroll
    for (int nf = 0; nf < 4; ++nf) {
      v16h b0 = frag_b(&Ks[(nf * 16) * 64], lane, 64);
      v16h b1 = frag_b(&Ks[(nf * 16) * 64 + 32], lane, 64);
      v8f z = {};
      z = wmma_f16(qa0, b0, z);
      s[nf] = wmma_f16(qa1, b1, z);
    }
#pragma unroll
    for (int nf = 0; nf < 4; ++nf)
#pragma unroll
      for (int i = 0; i < 8; ++i)
        Ssm[wave][(i + 8 * hf) * 64 + nf * 16 + ln] = s[nf][i];
    __syncthreads();

    // online softmax: lane handles row ln, columns hf*32..hf*32+31
    float pm = -1e30f;
#pragma unroll
    for (int c = 0; c < 32; ++c) pm = fmaxf(pm, Srow[c]);
    pm = fmaxf(pm, bpermf(pm, lane ^ 16));
    pm *= scale;
    float mnew = fmaxf(m, pm);
    float alpha = __expf(m - mnew);
    float sum = 0.f;
#pragma unroll
    for (int c = 0; c < 32; ++c) {
      float e = __expf(Srow[c] * scale - mnew);
      sum += e;
      Prow[c] = (_Float16)e;
    }
    sum += bpermf(sum, lane ^ 16);
    l = l * alpha + sum;
    m = mnew;
    __syncthreads();

    v16h pa0 = frag_a(&Psm[wave][0], lane, 64);   // keys 0..31
    v16h pa1 = frag_a(&Psm[wave][32], lane, 64);  // keys 32..63
#pragma unroll
    for (int i = 0; i < 8; ++i) {
      float af = bpermf(alpha, i + 8 * hf);
#pragma unroll
      for (int nf = 0; nf < 4; ++nf) o[nf][i] *= af;
    }
#pragma unroll
    for (int nf = 0; nf < 4; ++nf) {  // O += P . V, reduce over key
      v16h b0 = frag_b(&Vts[(nf * 16) * LDV], lane, LDV);
      v16h b1 = frag_b(&Vts[(nf * 16) * LDV + 32], lane, LDV);
      o[nf] = wmma_f16(pa0, b0, o[nf]);
      o[nf] = wmma_f16(pa1, b1, o[nf]);
    }
  }

  // normalize and store to attn buffer [B, N, H*64] as f16
  float linv = 1.f / l;
#pragma unroll
  for (int i = 0; i < 8; ++i) {
    float li = bpermf(linv, i + 8 * hf);
    int qg = qrow0 + i + 8 * hf;
    size_t rowoff = ((size_t)(b * 2048 + qg)) * 512 + h * 64;
#pragma unroll
    for (int nf = 0; nf < 4; ++nf)
      Oh[rowoff + nf * 16 + ln] = (_Float16)(o[nf][i] * li);
  }
}

// ------------- out: d_out[8192,512] = attn[8192,512] @ Wout + bout -----------
__global__ __launch_bounds__(256) void out_kernel(
    const _Float16* __restrict__ A, const float* __restrict__ Wout,
    const float* __restrict__ bout, float* __restrict__ Out) {
  __shared__ __align__(16) _Float16 As[128 * LDA];
  __shared__ _Float16 Bt[128 * LDB];
  const int t = threadIdx.x, lane = t & 31, wave = t >> 5;
  const int wm = wave & 3, wn = wave >> 2;
  const int rowBase = blockIdx.x * 128;
  const int colBase = blockIdx.y * 128;
  const unsigned long long abase_u = (unsigned long long)A;
  v8f acc[2][4] = {};
  for (int kt = 0; kt < 16; ++kt) {
    __syncthreads();
    {  // async DMA: A tile rows are pure f16 copies (32B per thread)
      int row = t >> 1, c0 = (t & 1) * 16;
      unsigned dst = lds_off(&As[row * LDA + c0]);
      unsigned voff =
          (unsigned)((((rowBase + row) * 512) + kt * 32 + c0) * 2);
      async_b128(dst, abase_u, voff);
      async_b128(dst + 16, abase_u, voff + 16);
    }
    stage_bt<128>(Bt, Wout, 512, colBase, kt * 32, t);
    if (kt + 1 < 16)
      __builtin_prefetch(A + (size_t)(rowBase + (t >> 1)) * 512 + kt * 32 + 32,
                         0, 0);
    wait_async();
    __syncthreads();
    v16h a0 = frag_a(As + (wm * 32) * LDA, lane, LDA);
    v16h a1 = frag_a(As + (wm * 32 + 16) * LDA, lane, LDA);
#pragma unroll
    for (int nf = 0; nf < 4; ++nf) {
      v16h b = frag_b(Bt + (wn * 64 + nf * 16) * LDB, lane, LDB);
      acc[0][nf] = wmma_f16(a0, b, acc[0][nf]);
      acc[1][nf] = wmma_f16(a1, b, acc[1][nf]);
    }
  }
  const int hf = lane >> 4, ln = lane & 15;
#pragma unroll
  for (int mf = 0; mf < 2; ++mf)
#pragma unroll
    for (int nf = 0; nf < 4; ++nf)
#pragma unroll
      for (int i = 0; i < 8; ++i) {
        int row = rowBase + wm * 32 + mf * 16 + i + 8 * hf;
        int col = colBase + wn * 64 + nf * 16 + ln;
        Out[(size_t)row * 512 + col] = acc[mf][nf][i] + bout[col];
      }
}

extern "C" void kernel_launch(void* const* d_in, const int* in_sizes, int n_in,
                              void* d_out, int out_size, void* d_ws,
                              size_t ws_size, hipStream_t stream) {
  (void)in_sizes; (void)n_in; (void)out_size; (void)ws_size;
  const float* x    = (const float*)d_in[0];
  const float* y    = (const float*)d_in[1];
  const float* Wqkv = (const float*)d_in[2];
  const float* Wfc  = (const float*)d_in[3];
  const float* bfc  = (const float*)d_in[4];
  const float* Wout = (const float*)d_in[5];
  const float* bout = (const float*)d_in[6];
  float* out = (float*)d_out;

  char* ws = (char*)d_ws;
  float* oq  = (float*)ws; ws += (size_t)8192 * 64 * 4;
  float* okv = (float*)ws; ws += (size_t)8192 * 64 * 4;
  _Float16* Qh = (_Float16*)ws; ws += (size_t)8192 * 512 * 2;
  _Float16* Kh = (_Float16*)ws; ws += (size_t)8192 * 512 * 2;
  _Float16* Vh = (_Float16*)ws; ws += (size_t)8192 * 512 * 2;
  _Float16* Ah = (_Float16*)ws; ws += (size_t)8192 * 512 * 2;

  dim3 blk(256);
  fc_kernel<<<dim3(64), blk, 0, stream>>>(x, Wfc, bfc, oq);
  fc_kernel<<<dim3(64), blk, 0, stream>>>(y, Wfc, bfc, okv);
  qkv_kernel<<<dim3(64, 12), blk, 0, stream>>>(x, Wqkv, oq, okv, Qh, Kh, Vh);
  attn_kernel<<<dim3(16, 8, 4), blk, 0, stream>>>(Qh, Kh, Vh, Ah);
  out_kernel<<<dim3(64, 4), blk, 0, stream>>>(Ah, Wout, bout, out);
}
